// SentSeq_43404939493661
// MI455X (gfx1250) — compile-verified, matrix operand order checked by reading, not run on hardware
//
#include <hip/hip_runtime.h>
#include <math.h>

// ---------------------------------------------------------------------------
// BERT-base forward on gfx1250 (CDNA5): bf16 WMMA GEMMs + async-to-LDS staging
// ---------------------------------------------------------------------------

#define DEVINL __device__ __forceinline__

typedef __attribute__((ext_vector_type(16))) __bf16 bf16x16;
typedef __attribute__((ext_vector_type(8)))  __bf16 bf16x8;
typedef __attribute__((ext_vector_type(8)))  float  f32x8;

constexpr int NB = 8;            // batch
constexpr int SQ = 512;          // sequence
constexpr int DM = 768;          // model dim
constexpr int NL = 12;           // layers
constexpr int NH = 12;           // heads
constexpr int FF = 3072;         // ffn dim
constexpr int DH = 64;           // head dim
constexpr int MM = NB * SQ;      // 4096 rows
constexpr int QLD = 3 * DM;      // qkv row stride
constexpr float QSCALE = 0.125f; // 1/sqrt(64)
constexpr float LNEPS  = 1e-12f;

DEVINL f32x8 wmma_bf16(bf16x16 a, bf16x16 b, f32x8 c) {
  // D = A(16x32 bf16) * B(32x16 bf16) + C(16x16 f32)
  return __builtin_amdgcn_wmma_f32_16x16x32_bf16(false, a, false, b, (short)0, c, false, false);
}

// CDNA5 async copy: global -> LDS, 16B per lane, tracked by ASYNCcnt.
DEVINL void async_load_b128(unsigned lds_off, const void* gptr) {
  unsigned long long ga = (unsigned long long)(uintptr_t)gptr;
  asm volatile("global_load_async_to_lds_b128 %0, %1, off"
               :: "v"(lds_off), "v"(ga) : "memory");
}
DEVINL void wait_async0() {
  asm volatile("s_wait_asynccnt 0x0" ::: "memory");
}

// Fragment loader for 16-bit A/B fragments from a row-major [16+][K] region.
// lane<16 : row = lane,    k elements {0..7, 16..23}
// lane>=16: row = lane-16, k elements {8..15, 24..31}
DEVINL bf16x16 load_frag(const __bf16* base, int ld, int lane) {
  const int r    = lane & 15;
  const int koff = (lane >> 4) << 3;
  const __bf16* p = base + r * ld + koff;
  bf16x8 lo = *(const bf16x8*)(p);
  bf16x8 hi = *(const bf16x8*)(p + 16);
  return __builtin_shufflevector(lo, hi, 0,1,2,3,4,5,6,7,8,9,10,11,12,13,14,15);
}

// ---------------------------------------------------------------------------
// Tiled WMMA GEMM:  C[M][Nc] = A[M][K](bf16) * Bt[Nc][K](bf16)^T + bias
// Block tile 128x128, 8 waves, each wave 64(m) x 32(n) = 4x2 wmma tiles.
// Tiles staged to LDS with global_load_async_to_lds_b128 (ASYNCcnt).
// EPI 0: fp32 out.  EPI 1: gelu -> bf16 out.  EPI 2: qkv (scale Q, write V^T).
// ---------------------------------------------------------------------------
template<int EPI>
__global__ __launch_bounds__(256) void gemm_wmma(
    const __bf16* __restrict__ A, const __bf16* __restrict__ Bt,
    const float* __restrict__ bias, int K, int Nc,
    float* __restrict__ outf, __bf16* __restrict__ outb, __bf16* __restrict__ vt)
{
  __shared__ __bf16 As[128][48];   // 96B row stride: 16B aligned, bank-spread
  __shared__ __bf16 Bs[128][48];
  const int tid = threadIdx.x, lane = tid & 31, w = tid >> 5;
  const int wm = w & 1, wn = w >> 1;
  const int m0 = blockIdx.y * 128, n0 = blockIdx.x * 128;

  // LDS byte offsets (generic LDS pointer low 32 bits == LDS address)
  const unsigned asBase = (unsigned)(uintptr_t)&As[0][0];
  const unsigned bsBase = (unsigned)(uintptr_t)&Bs[0][0];

  f32x8 acc[4][2] = {};

  for (int k0 = 0; k0 < K; k0 += 32) {
    __syncthreads();                     // previous compute done before overwrite
    #pragma unroll
    for (int i = 0; i < 2; ++i) {
      int qq  = tid + i * 256;           // 512 chunks of 8 bf16 per matrix
      int row = qq >> 2;
      int kc  = (qq & 3) << 3;
      unsigned loff = (unsigned)(row * 48 + kc) * 2u;
      async_load_b128(asBase + loff, A  + (size_t)(m0 + row) * K + k0 + kc);
      async_load_b128(bsBase + loff, Bt + (size_t)(n0 + row) * K + k0 + kc);
      if (k0 + 32 < K) {                 // hint next K-tile -> global_prefetch_b8
        __builtin_prefetch(A  + (size_t)(m0 + row) * K + k0 + 32 + kc, 0, 1);
        __builtin_prefetch(Bt + (size_t)(n0 + row) * K + k0 + 32 + kc, 0, 1);
      }
    }
    wait_async0();                       // this wave's async copies landed in LDS
    __syncthreads();                     // all waves' copies visible

    bf16x16 af[4], bf[2];
    #pragma unroll
    for (int tm = 0; tm < 4; ++tm) af[tm] = load_frag(&As[wm * 64 + tm * 16][0], 48, lane);
    #pragma unroll
    for (int tn = 0; tn < 2; ++tn) bf[tn] = load_frag(&Bs[wn * 32 + tn * 16][0], 48, lane);
    #pragma unroll
    for (int tm = 0; tm < 4; ++tm)
      #pragma unroll
      for (int tn = 0; tn < 2; ++tn)
        acc[tm][tn] = wmma_bf16(af[tm], bf[tn], acc[tm][tn]);
  }

  // Epilogue.  C layout: lane<16 -> (m=r, n=lane); lane>=16 -> (m=8+r, n=lane-16)
  const int col = lane & 15, half = lane >> 4;
  #pragma unroll
  for (int tm = 0; tm < 4; ++tm) {
    #pragma unroll
    for (int tn = 0; tn < 2; ++tn) {
      const int n  = n0 + wn * 32 + tn * 16 + col;
      const int mb = m0 + wm * 64 + tm * 16 + half * 8;
      const float bv = bias[n];
      #pragma unroll
      for (int r = 0; r < 8; ++r) {
        const int m = mb + r;
        float v = acc[tm][tn][r] + bv;
        if (EPI == 0) {
          outf[(size_t)m * Nc + n] = v;
        } else if (EPI == 1) {
          float gl = 0.5f * v * (1.0f + erff(v * 0.70710678118f));
          outb[(size_t)m * Nc + n] = (__bf16)gl;
        } else {
          float sv = (n < DM) ? v * QSCALE : v;   // fold 1/sqrt(DH) into Q
          outb[(size_t)m * Nc + n] = (__bf16)sv;
          if (n >= 2 * DM) {                      // V -> V^T[b][h][dh][s]
            int nh = n - 2 * DM, hh = nh >> 6, dh = nh & 63;
            int b = m >> 9, s = m & 511;
            vt[(size_t)((b * NH + hh) * DH + dh) * SQ + s] = (__bf16)v;
          }
        }
      }
    }
  }
}

// ---------------------------------------------------------------------------
// Flash-style attention.  Block = (b,h, 128 queries); each wave owns 16 queries.
// S^T = K * Q^T  (C layout: lane = query column, rows = keys)
// O^T += V^T * P^T  (P^T accumulator -> B-fragment with NO cross-lane movement)
// ---------------------------------------------------------------------------
__global__ __launch_bounds__(256) void attention_wmma(
    const __bf16* __restrict__ qkv, const __bf16* __restrict__ vt,
    const float* __restrict__ biasrow, __bf16* __restrict__ attn)
{
  const int lane = threadIdx.x & 31, w = threadIdx.x >> 5;
  const int bh = blockIdx.y, b = bh / NH, h = bh % NH;
  const int q0 = blockIdx.x * 128 + w * 16;
  const int koff = (lane >> 4) << 3;

  const bf16x16 qf0 = load_frag(qkv + (size_t)(b * SQ + q0) * QLD + h * DH +  0, QLD, lane);
  const bf16x16 qf1 = load_frag(qkv + (size_t)(b * SQ + q0) * QLD + h * DH + 32, QLD, lane);

  f32x8 o[4] = {};
  const f32x8 zero = {};
  float mrun = -1e30f, lsum = 0.0f;

  for (int s0 = 0; s0 < SQ; s0 += 32) {
    f32x8 st[2];
    #pragma unroll
    for (int sm = 0; sm < 2; ++sm) {
      const __bf16* kb = qkv + (size_t)(b * SQ + s0 + sm * 16) * QLD + DM + h * DH;
      f32x8 c = wmma_bf16(load_frag(kb,      QLD, lane), qf0, zero);
      st[sm]  = wmma_bf16(load_frag(kb + 32, QLD, lane), qf1, c);
    }
    const float* bp = biasrow + b * SQ + s0 + koff;
    float tmax = -1e30f;
    #pragma unroll
    for (int r = 0; r < 8; ++r) {
      st[0][r] += bp[r];
      st[1][r] += bp[16 + r];
      tmax = fmaxf(tmax, fmaxf(st[0][r], st[1][r]));
    }
    tmax = fmaxf(tmax, __shfl_xor(tmax, 16));       // combine lane halves (same q)
    const float mnew = fmaxf(mrun, tmax);
    const float corr = __expf(mrun - mnew);
    #pragma unroll
    for (int dt = 0; dt < 4; ++dt) o[dt] *= corr;
    lsum *= corr;

    bf16x16 pb;
    float psum = 0.0f;
    #pragma unroll
    for (int r = 0; r < 8; ++r) {
      float p0 = __expf(st[0][r] - mnew);
      float p1 = __expf(st[1][r] - mnew);
      psum += p0 + p1;
      pb[r]     = (__bf16)p0;   // tile s=0..15  -> B-frag k elems 0..7
      pb[8 + r] = (__bf16)p1;   // tile s=16..31 -> B-frag k elems 8..15
    }
    psum += __shfl_xor(psum, 16);
    lsum += psum;
    mrun = mnew;

    #pragma unroll
    for (int dt = 0; dt < 4; ++dt) {
      bf16x16 va = load_frag(vt + (size_t)((b * NH + h) * DH + dt * 16) * SQ + s0, SQ, lane);
      o[dt] = wmma_bf16(va, pb, o[dt]);
    }
  }

  const float inv = 1.0f / lsum;
  const int q = q0 + (lane & 15);
  #pragma unroll
  for (int dt = 0; dt < 4; ++dt) {
    bf16x8 ov;
    #pragma unroll
    for (int r = 0; r < 8; ++r) ov[r] = (__bf16)(o[dt][r] * inv);
    *(bf16x8*)(attn + (size_t)(b * SQ + q) * DM + h * DH + dt * 16 + koff) = ov;
  }
}

// ---------------------------------------------------------------------------
// Row reductions / LayerNorm
// ---------------------------------------------------------------------------
DEVINL float block_sum(float v) {
  __shared__ float sh[256];
  const int t = threadIdx.x;
  sh[t] = v; __syncthreads();
  #pragma unroll
  for (int o = 128; o > 0; o >>= 1) { if (t < o) sh[t] += sh[t + o]; __syncthreads(); }
  const float r = sh[0]; __syncthreads();
  return r;
}

__global__ __launch_bounds__(256) void embed_ln_kernel(
    const int* __restrict__ ids, const float* __restrict__ tok,
    const float* __restrict__ pos, const float* __restrict__ g,
    const float* __restrict__ bta, float* __restrict__ xf, __bf16* __restrict__ xb)
{
  const int m = blockIdx.x, s = m & (SQ - 1);
  const int id = ids[m];
  float v[3]; float s1 = 0.f, s2 = 0.f;
  #pragma unroll
  for (int j = 0; j < 3; ++j) {
    const int d = threadIdx.x + j * 256;
    v[j] = tok[(size_t)id * DM + d] + pos[(size_t)s * DM + d];
    s1 += v[j]; s2 += v[j] * v[j];
  }
  s1 = block_sum(s1); s2 = block_sum(s2);
  const float mean = s1 * (1.0f / DM);
  const float var  = s2 * (1.0f / DM) - mean * mean;
  const float rstd = rsqrtf(var + LNEPS);
  #pragma unroll
  for (int j = 0; j < 3; ++j) {
    const int d = threadIdx.x + j * 256;
    const float y = (v[j] - mean) * rstd * g[d] + bta[d];
    xf[(size_t)m * DM + d] = y;
    xb[(size_t)m * DM + d] = (__bf16)y;
  }
}

__global__ __launch_bounds__(256) void ln_res_kernel(
    const float* __restrict__ res, const float* __restrict__ tmp,
    const float* __restrict__ g, const float* __restrict__ bta,
    float* __restrict__ yf, __bf16* __restrict__ yb)
{
  const int m = blockIdx.x;
  float v[3]; float s1 = 0.f, s2 = 0.f;
  #pragma unroll
  for (int j = 0; j < 3; ++j) {
    const int d = threadIdx.x + j * 256;
    v[j] = res[(size_t)m * DM + d] + tmp[(size_t)m * DM + d];
    s1 += v[j]; s2 += v[j] * v[j];
  }
  s1 = block_sum(s1); s2 = block_sum(s2);
  const float mean = s1 * (1.0f / DM);
  const float var  = s2 * (1.0f / DM) - mean * mean;
  const float rstd = rsqrtf(var + LNEPS);
  #pragma unroll
  for (int j = 0; j < 3; ++j) {
    const int d = threadIdx.x + j * 256;
    const float y = (v[j] - mean) * rstd * g[d] + bta[d];
    yf[(size_t)m * DM + d] = y;
    yb[(size_t)m * DM + d] = (__bf16)y;
  }
}

// ---------------------------------------------------------------------------
// Small helpers
// ---------------------------------------------------------------------------
__global__ void biasrow_kernel(const int* __restrict__ amask, float* __restrict__ br) {
  int i = blockIdx.x * 256 + threadIdx.x;
  if (i < NB * SQ) br[i] = (1.0f - (float)amask[i]) * -1e9f;
}

// fp32 [K][Nc] -> bf16 [Nc][K]  (weight transpose-convert, once per layer/call)
__global__ void convT_kernel(const float* __restrict__ W, __bf16* __restrict__ Wt, int K, int Nc) {
  int i = blockIdx.x * 256 + threadIdx.x;
  if (i < K * Nc) {
    int k = i / Nc, n = i - k * Nc;
    Wt[(size_t)n * K + k] = (__bf16)W[i];
  }
}

// Ragged gather/scatter of [SEP]=102 / [MASK]=103 rows -> out[b][0..63][D]
__global__ __launch_bounds__(256) void gather_scatter_kernel(
    const int* __restrict__ ids, const float* __restrict__ hsrc, float* __restrict__ out)
{
  const int b = blockIdx.x;
  __shared__ int dest[SQ];
  if (threadIdx.x == 0) {
    int ncls = 0, nsent = 0;
    for (int s = 0; s < SQ; ++s) {
      int id = ids[b * SQ + s];
      ncls  += (id == 102);
      nsent += (id == 103);
    }
    int rc = 0, rs = 0;
    for (int s = 0; s < SQ; ++s) {
      int id = ids[b * SQ + s];
      int d = 64; bool sel = false;
      if (id == 102)      { d = rc++;        sel = true; }
      else if (id == 103) { d = ncls + rs++; sel = true; }
      bool ok = sel && (d < nsent) && (d < 64);
      dest[s] = ok ? d : -1;
    }
  }
  __syncthreads();
  for (int i = threadIdx.x; i < 64 * DM; i += 256)
    out[(size_t)b * 64 * DM + i] = 0.0f;
  __syncthreads();
  for (int s = 0; s < SQ; ++s) {
    int d = dest[s];
    if (d >= 0)
      for (int i = threadIdx.x; i < DM; i += 256)
        out[((size_t)b * 64 + d) * DM + i] = hsrc[((size_t)b * SQ + s) * DM + i];
  }
}

// ---------------------------------------------------------------------------
// Orchestration
// ---------------------------------------------------------------------------
extern "C" void kernel_launch(void* const* d_in, const int* in_sizes, int n_in,
                              void* d_out, int out_size, void* d_ws, size_t ws_size,
                              hipStream_t stream) {
  (void)in_sizes; (void)n_in; (void)out_size; (void)ws_size;
  const int*   ids   = (const int*)  d_in[0];
  const int*   amask = (const int*)  d_in[1];
  const float* tok   = (const float*)d_in[2];
  const float* pos   = (const float*)d_in[3];
  const float* elng  = (const float*)d_in[4];
  const float* elnb  = (const float*)d_in[5];
  const float* Wqkv  = (const float*)d_in[6];
  const float* bqkv  = (const float*)d_in[7];
  const float* Wo    = (const float*)d_in[8];
  const float* bo    = (const float*)d_in[9];
  const float* ln1g  = (const float*)d_in[10];
  const float* ln1b  = (const float*)d_in[11];
  const float* W1    = (const float*)d_in[12];
  const float* b1    = (const float*)d_in[13];
  const float* W2    = (const float*)d_in[14];
  const float* b2    = (const float*)d_in[15];
  const float* ln2g  = (const float*)d_in[16];
  const float* ln2b  = (const float*)d_in[17];

  char* p = (char*)d_ws;
  auto take = [&](size_t bytes) { char* r = p; p += (bytes + 255) & ~(size_t)255; return r; };

  float*  xf   = (float*) take((size_t)MM * DM * 4);
  __bf16* xb   = (__bf16*)take((size_t)MM * DM * 2);
  float*  x2f  = (float*) take((size_t)MM * DM * 4);
  __bf16* x2b  = (__bf16*)take((size_t)MM * DM * 2);
  float*  tmpf = (float*) take((size_t)MM * DM * 4);
  __bf16* qkvb = (__bf16*)take((size_t)MM * 3 * DM * 2);
  __bf16* vtb  = (__bf16*)take((size_t)MM * DM * 2);
  __bf16* attb = (__bf16*)take((size_t)MM * DM * 2);
  __bf16* ffhb = (__bf16*)take((size_t)MM * FF * 2);
  __bf16* wqkT = (__bf16*)take((size_t)DM * 3 * DM * 2);
  __bf16* woT  = (__bf16*)take((size_t)DM * DM * 2);
  __bf16* w1T  = (__bf16*)take((size_t)DM * FF * 2);
  __bf16* w2T  = (__bf16*)take((size_t)FF * DM * 2);
  float*  brow = (float*) take((size_t)NB * SQ * 4);

  dim3 blk(256);
  biasrow_kernel<<<(NB * SQ + 255) / 256, blk, 0, stream>>>(amask, brow);
  embed_ln_kernel<<<MM, blk, 0, stream>>>(ids, tok, pos, elng, elnb, xf, xb);

  for (int l = 0; l < NL; ++l) {
    convT_kernel<<<(DM * 3 * DM + 255) / 256, blk, 0, stream>>>(Wqkv + (size_t)l * DM * 3 * DM, wqkT, DM, 3 * DM);
    gemm_wmma<2><<<dim3(3 * DM / 128, MM / 128), blk, 0, stream>>>(
        xb, wqkT, bqkv + (size_t)l * 3 * DM, DM, 3 * DM, nullptr, qkvb, vtb);

    attention_wmma<<<dim3(SQ / 128, NB * NH), blk, 0, stream>>>(qkvb, vtb, brow, attb);

    convT_kernel<<<(DM * DM + 255) / 256, blk, 0, stream>>>(Wo + (size_t)l * DM * DM, woT, DM, DM);
    gemm_wmma<0><<<dim3(DM / 128, MM / 128), blk, 0, stream>>>(
        attb, woT, bo + (size_t)l * DM, DM, DM, tmpf, nullptr, nullptr);
    ln_res_kernel<<<MM, blk, 0, stream>>>(xf, tmpf, ln1g + (size_t)l * DM, ln1b + (size_t)l * DM, x2f, x2b);

    convT_kernel<<<(DM * FF + 255) / 256, blk, 0, stream>>>(W1 + (size_t)l * DM * FF, w1T, DM, FF);
    gemm_wmma<1><<<dim3(FF / 128, MM / 128), blk, 0, stream>>>(
        x2b, w1T, b1 + (size_t)l * FF, DM, FF, nullptr, ffhb, nullptr);

    convT_kernel<<<(FF * DM + 255) / 256, blk, 0, stream>>>(W2 + (size_t)l * FF * DM, w2T, FF, DM);
    gemm_wmma<0><<<dim3(DM / 128, MM / 128), blk, 0, stream>>>(
        ffhb, w2T, b2 + (size_t)l * DM, FF, DM, tmpf, nullptr, nullptr);
    ln_res_kernel<<<MM, blk, 0, stream>>>(x2f, tmpf, ln2g + (size_t)l * DM, ln2b + (size_t)l * DM, xf, xb);
  }

  gather_scatter_kernel<<<NB, blk, 0, stream>>>(ids, xf, (float*)d_out);
}
